// NodeModel_54589034332475
// MI455X (gfx1250) — matrix operand order repaired and checked by reading, not compiled, not run on hardware
//
#include <hip/hip_runtime.h>

#define N_NODES 50000
#define N_EDGES 800000
#define FX      128
#define FE      128
#define IN_DIM  256
#define OUT_DIM 128
#define N_TILES_NODE (N_NODES / 16)   // 3125, exact
#define N_TILES_EDGE (N_EDGES / 16)   // 50000, exact

typedef __attribute__((ext_vector_type(16))) __bf16 v16bf;
typedef __attribute__((ext_vector_type(16))) float  v16f;
typedef __attribute__((ext_vector_type(8)))  float  v8f;

// Stage a [IN_DIM][OUT_DIM] fp32 weight matrix into LDS as bf16 B-fragments.
// Fragment (nt,kt) is the 32x16 slice K=[kt*32,kt*32+32), N=[nt*16,nt*16+16).
// B layout (16-bit, 32x16): lanes 0-15 -> column N=lane, K offsets {0..7, 16..23};
// lanes 16-31 -> column N=lane-16, K offsets {8..15, 24..31}; element j within a
// lane's 16-value run maps to increasing K (pairs pack into VGPRs naturally).
__device__ __forceinline__ void stage_weights(const float* __restrict__ W,
                                              __bf16* lds) {
  for (int idx = threadIdx.x; idx < 8 * 8 * 32 * 16; idx += blockDim.x) {
    int j    = idx & 15;
    int lane = (idx >> 4) & 31;
    int kt   = (idx >> 9) & 7;
    int nt   = (idx >> 12) & 7;
    int K = kt * 32 + ((j >> 3) << 4) + ((lane >> 4) << 3) + (j & 7);
    int n = nt * 16 + (lane & 15);
    lds[idx] = (__bf16)W[K * OUT_DIM + n];
  }
}

// Build one 16x32 bf16 A-fragment for this lane from a row of 32 fp32 values.
// 16-bit A layout: lanes 0-15 hold K {0..7, 16..23}; lanes 16-31 hold
// K {8..15, 24..31}; row M = lane & 15.
__device__ __forceinline__ v16bf build_a_frag(const float* __restrict__ src,
                                              int khalf, float scale) {
  const float4 f0 = *(const float4*)(src + khalf);
  const float4 f1 = *(const float4*)(src + khalf + 4);
  const float4 f2 = *(const float4*)(src + khalf + 16);
  const float4 f3 = *(const float4*)(src + khalf + 20);
  v16f fv;
  fv.s0 = f0.x; fv.s1 = f0.y; fv.s2 = f0.z; fv.s3 = f0.w;
  fv.s4 = f1.x; fv.s5 = f1.y; fv.s6 = f1.z; fv.s7 = f1.w;
  fv.s8 = f2.x; fv.s9 = f2.y; fv.sa = f2.z; fv.sb = f2.w;
  fv.sc = f3.x; fv.sd = f3.y; fv.se = f3.z; fv.sf = f3.w;
  fv *= scale;
  return __builtin_convertvector(fv, v16bf);   // native packed f32->bf16 cvt
}

__global__ void zero_kernel(float* __restrict__ out, float* __restrict__ cnt) {
  int i = blockIdx.x * blockDim.x + threadIdx.x;
  if (i < N_NODES * OUT_DIM) out[i] = 0.0f;
  if (i < N_NODES) cnt[i] = 0.0f;
}

// Edge MLP: per 16-edge tile, data = relu(concat(x[row], edge_attr) @ W1 + b1)
// then atomic scatter-add into summed[col] and count[col].
__global__ void __launch_bounds__(256)
edge_kernel(const float* __restrict__ x, const int* __restrict__ ei,
            const float* __restrict__ ea, const float* __restrict__ W1,
            const float* __restrict__ b1, float* __restrict__ summed,
            float* __restrict__ cnt) {
  __shared__ __bf16 wlds[8 * 8 * 32 * 16];  // 64 KB
  stage_weights(W1, wlds);
  __syncthreads();

  const int lane  = threadIdx.x & 31;
  const int wave  = threadIdx.x >> 5;
  const int tile  = blockIdx.x * 8 + wave;          // exactly N_TILES_EDGE waves
  const int base  = tile * 16;
  const int m     = lane & 15;
  const int hi    = lane >> 4;                      // 0 or 1
  const int e     = base + m;
  const int khalf = hi * 8;
  const int srcRow = ei[e];                         // edge_index[0][e]

  v8f acc[8];
#pragma unroll
  for (int nt = 0; nt < 8; ++nt)
#pragma unroll
    for (int r = 0; r < 8; ++r) acc[nt][r] = 0.0f;

#pragma unroll
  for (int kt = 0; kt < 8; ++kt) {
    const float* src = (kt < 4)
        ? (x + (size_t)srcRow * FX + kt * 32)
        : (ea + (size_t)e * FE + (kt - 4) * 32);
    const v16bf a = build_a_frag(src, khalf, 1.0f);
#pragma unroll
    for (int nt = 0; nt < 8; ++nt) {
      const v16bf b = *(const v16bf*)&wlds[((nt * 8 + kt) * 32 + lane) * 16];
      acc[nt] = __builtin_amdgcn_wmma_f32_16x16x32_bf16(
          false, a, false, b, (short)0, acc[nt], false, false);
    }
  }

  // destination node for each C row this lane owns (M = r + hi*8)
  int cols[8];
#pragma unroll
  for (int r = 0; r < 8; ++r) cols[r] = ei[N_EDGES + base + hi * 8 + r];

  const int n0 = lane & 15;
#pragma unroll
  for (int nt = 0; nt < 8; ++nt) {
    const int n = nt * 16 + n0;
    const float bias = b1[n];
#pragma unroll
    for (int r = 0; r < 8; ++r) {
      float v = acc[nt][r] + bias;
      v = v > 0.0f ? v : 0.0f;
      atomicAdd(summed + (size_t)cols[r] * OUT_DIM + n, v);
    }
  }
  if (lane < 16) atomicAdd(cnt + ei[N_EDGES + e], 1.0f);
}

// Node MLP: per 16-node tile, out = relu(concat(x, summed/max(cnt,1)) @ W2 + b2).
// `out` holds the scatter-sums on entry; each wave reads only its own 16 rows
// (all reads precede the epilogue stores), then overwrites them.
__global__ void __launch_bounds__(256)
node_kernel(const float* __restrict__ x, const float* __restrict__ W2,
            const float* __restrict__ b2, const float* __restrict__ cnt,
            float* __restrict__ out) {
  __shared__ __bf16 wlds[8 * 8 * 32 * 16];  // 64 KB
  stage_weights(W2, wlds);
  __syncthreads();

  const int lane = threadIdx.x & 31;
  const int wave = threadIdx.x >> 5;
  const int tile = blockIdx.x * 8 + wave;
  if (tile >= N_TILES_NODE) return;
  const int base  = tile * 16;
  const int m     = lane & 15;
  const int hi    = lane >> 4;
  const int node  = base + m;
  const int khalf = hi * 8;

  const float c   = cnt[node];
  const float inv = 1.0f / (c > 1.0f ? c : 1.0f);

  v8f acc[8];
#pragma unroll
  for (int nt = 0; nt < 8; ++nt)
#pragma unroll
    for (int r = 0; r < 8; ++r) acc[nt][r] = 0.0f;

#pragma unroll
  for (int kt = 0; kt < 8; ++kt) {
    const float* src = (kt < 4)
        ? (x + (size_t)node * FX + kt * 32)
        : (out + (size_t)node * OUT_DIM + (kt - 4) * 32);
    const float scale = (kt < 4) ? 1.0f : inv;      // agg = summed / max(cnt,1)
    const v16bf a = build_a_frag(src, khalf, scale);
#pragma unroll
    for (int nt = 0; nt < 8; ++nt) {
      const v16bf b = *(const v16bf*)&wlds[((nt * 8 + kt) * 32 + lane) * 16];
      acc[nt] = __builtin_amdgcn_wmma_f32_16x16x32_bf16(
          false, a, false, b, (short)0, acc[nt], false, false);
    }
  }

  const int n0 = lane & 15;
#pragma unroll
  for (int nt = 0; nt < 8; ++nt) {
    const int n = nt * 16 + n0;
    const float bias = b2[n];
#pragma unroll
    for (int r = 0; r < 8; ++r) {
      const int row = base + hi * 8 + r;
      float v = acc[nt][r] + bias;
      v = v > 0.0f ? v : 0.0f;
      out[(size_t)row * OUT_DIM + n] = v;
    }
  }
}

extern "C" void kernel_launch(void* const* d_in, const int* in_sizes, int n_in,
                              void* d_out, int out_size, void* d_ws, size_t ws_size,
                              hipStream_t stream) {
  const float* x  = (const float*)d_in[0];
  const int*   ei = (const int*)d_in[1];   // [2, E] flat
  const float* ea = (const float*)d_in[2];
  // d_in[3] = u, d_in[4] = batch : unused by the reference computation
  const float* W1 = (const float*)d_in[5];
  const float* b1 = (const float*)d_in[6];
  const float* W2 = (const float*)d_in[7];
  const float* b2 = (const float*)d_in[8];

  float* out = (float*)d_out;              // doubles as the scatter-sum buffer
  float* cnt = (float*)d_ws;               // N_NODES floats

  zero_kernel<<<(N_NODES * OUT_DIM + 255) / 256, 256, 0, stream>>>(out, cnt);
  edge_kernel<<<N_TILES_EDGE / 8, 256, 0, stream>>>(x, ei, ea, W1, b1, out, cnt);
  node_kernel<<<(N_TILES_NODE + 7) / 8, 256, 0, stream>>>(x, W2, b2, cnt, out);
}
